// RNNBase_79620103733955
// MI455X (gfx1250) — matrix operand order repaired and checked by reading, not compiled, not run on hardware
//
#include <hip/hip_runtime.h>
#include <cstdint>
#include <cstddef>

// ---------------- problem constants ----------------
#define T_SEQ   1024
#define BATCH   64
#define HID     512
#define GATES4  (4 * HID)        // 2048
#define NLAYER  2
#define WG_PER_DIR 16            // each WG owns 32 hidden units
#define NU      32               // hidden units per WG
#define NC      128              // gate columns per WG (4 gates x 32 units)

// ---------------- vector types ----------------
typedef __attribute__((ext_vector_type(16))) __bf16        v16bf;
typedef __attribute__((ext_vector_type(8)))  __bf16        v8bf;
typedef __attribute__((ext_vector_type(8)))  float         v8f;
typedef __attribute__((ext_vector_type(4)))  unsigned int  u32x4;
typedef __attribute__((ext_vector_type(4)))  int           i32x4;
typedef __attribute__((ext_vector_type(8)))  int           i32x8;

// ---------------- dynamic-LDS layout (bytes) ----------------
#define OFF_WX   0          // Wx slice [128 rows][512 k] bf16 : 131072
#define OFF_WH   131072     // Wh slice [128 rows][512 k] bf16 : 131072
#define OFF_G    262144     // gate staging [64 b][128 col] f32:  32768
#define OFF_C    294912     // cell state  [64 b][32 u]    f32 :   8192
#define OFF_B    303104     // bias slice  [128]           f32 :    512
#define SMEM_SZ  303616

// ---------------- workspace layout (bytes) ----------------
#define WS_WXBF  ((size_t)0)                         // [2 dir][2 lay][2048][512] bf16 = 8 MB
#define WS_WHBF  ((size_t)8388608)                   // same                         = 8 MB
#define WS_BIAS  ((size_t)16777216)                  // [2 dir][2 lay][2048] f32     = 32 KB
#define WS_HBUF  ((size_t)16809984)                  // [2 dir][2 slab][64][512] bf16= 256 KB
#define WS_BAR   ((size_t)17072128)                  // 16 ints
#define WS_XBF   ((size_t)17072192)                  // [T][B][H] bf16 = 64 MB
#define WS_SEQ0  ((size_t)84181056)                  // [2 dir][T][B][H] bf16 = 128 MB

__device__ __forceinline__ unsigned short f2bf(float f) {
  unsigned int u = __float_as_uint(f);
  u += 0x7FFFu + ((u >> 16) & 1u);                 // round-to-nearest-even
  return (unsigned short)(u >> 16);
}
__device__ __forceinline__ float sigm(float x) { return 1.0f / (1.0f + __expf(-x)); }

// A-fragment load: 16 contiguous K (two 16B chunks at K and K+8)
__device__ __forceinline__ v16bf ldA(const unsigned short* ap) {
  v8bf alo = *(const v8bf*)(ap);
  v8bf ahi = *(const v8bf*)(ap + 8);
  return __builtin_shufflevector(alo, ahi, 0, 1, 2, 3, 4, 5, 6, 7,
                                 8, 9, 10, 11, 12, 13, 14, 15);
}

// ---------------- helpers: fp32 -> bf16 bulk convert ----------------
__global__ void cvt_bf16_kernel(const float* __restrict__ s,
                                unsigned short* __restrict__ d, long n) {
  long i = (long)blockIdx.x * blockDim.x + threadIdx.x;
  long st = (long)gridDim.x * blockDim.x;
  for (; i < n; i += st) d[i] = f2bf(s[i]);
}

// combine bx+bh biases per (dir,layer) and zero the inter-WG barriers
__global__ void bias_bar_kernel(const float* __restrict__ bxf, const float* __restrict__ bhf,
                                const float* __restrict__ bxb, const float* __restrict__ bhb,
                                float* __restrict__ bias, int* __restrict__ bar) {
  int i = blockIdx.x * blockDim.x + threadIdx.x;            // 0..8191
  if (i < 2 * GATES4)            bias[i] = bxf[i] + bhf[i];             // dir 0, layers 0..1
  else if (i < 4 * GATES4) { int j = i - 2 * GATES4; bias[i] = bxb[j] + bhb[j]; }
  if (i < 16) bar[i] = 0;
}

// ---------------- inter-WG barrier (16 WGs of one direction) ----------------
__device__ __forceinline__ void dir_barrier(int* cnt, int* flg, int sense, int tid) {
  __syncthreads();
  if (tid == 0) {
    if (atomicAdd(cnt, 1) == WG_PER_DIR - 1) {
      atomicExch(cnt, 0);
      __threadfence();
      __hip_atomic_store(flg, sense, __ATOMIC_RELEASE, __HIP_MEMORY_SCOPE_AGENT);
    } else {
      while (__hip_atomic_load(flg, __ATOMIC_ACQUIRE, __HIP_MEMORY_SCOPE_AGENT) < sense) {}
    }
  }
  // CDNA5 cluster barrier (s_barrier_signal/wait -3): the hardware path when the
  // 16 WGs of a direction are dispatched as one workgroup-cluster.
  __builtin_amdgcn_s_cluster_barrier();
  __syncthreads();
}

// ---------------- TDM: 2D tile (32 rows x 512 bf16, row stride 512) -> LDS ----------------
// D# group0: {count=1, lds_addr, global_addr[31:0], global_addr[56:32] | type=2}
// D# group1 dwords: d0 data_size=1(2B)<<16; d1 tensor_dim0(512)<<16; d2 tensor_dim1(32)<<16;
//                   d3 tile_dim0(512)<<16; d4 tile_dim1(32); d5 tensor_dim0_stride(512).
__device__ __forceinline__ void tdm_load_block(const unsigned short* gsrc, unsigned lds_byte) {
  unsigned long long ga = (unsigned long long)(uintptr_t)gsrc;
  u32x4 g0 = { 1u, lds_byte, (unsigned)ga,
               (unsigned)((ga >> 32) & 0x1FFFFFFu) | (2u << 30) };
  i32x8 g1 = { (int)(1u << 16),
               (int)(512u << 16),
               (int)(32u  << 16),
               (int)(512u << 16),
               32, 512, 0, 0 };
  i32x4 z4 = {};
#if __clang_major__ >= 23
  i32x8 z8 = {};
  __builtin_amdgcn_tensor_load_to_lds(g0, g1, z4, z4, z8, 0);
#else
  __builtin_amdgcn_tensor_load_to_lds(g0, g1, z4, z4, 0);
#endif
}

// One GEMM pass: acc[4] += A(64x512 bf16, from `ab`) * B(512x128 slice in LDS `wb`)
// Software pipeline unrolled by 2: even k-tiles compute from buffer 0 while
// buffer 1 loads, odd k-tiles compute from buffer 1 while buffer 0 loads.
// No buffer-rotation copies; WMMAs cover the next tile's L2/LDS latency.
#define GEMM_PASS(AB, WB)                                                                   \
  {                                                                                         \
    v16bf a0[4], a1[4];                                                                     \
    v16bf b0, b1;                                                                           \
    const unsigned short* wrow = (WB) + (size_t)(wv * 16 + lh) * HID;                       \
    b0 = *(const v16bf*)(wrow + sel * 16);                                                  \
    _Pragma("unroll")                                                                       \
    for (int mt = 0; mt < 4; ++mt)                                                          \
      a0[mt] = ldA((AB) + (size_t)(mt * 16 + lh) * HID + sel * 16);                         \
    for (int kp = 0; kp < 8; ++kp) {                                                        \
      const int k1 = (2 * kp + 1) * 32;                                                     \
      b1 = *(const v16bf*)(wrow + k1 + sel * 16);                                           \
      _Pragma("unroll")                                                                     \
      for (int mt = 0; mt < 4; ++mt)                                                        \
        a1[mt] = ldA((AB) + (size_t)(mt * 16 + lh) * HID + k1 + sel * 16);                  \
      _Pragma("unroll")                                                                     \
      for (int mt = 0; mt < 4; ++mt)                                                        \
        acc[mt] = __builtin_amdgcn_wmma_f32_16x16x32_bf16(                                  \
            false, a0[mt], false, b0, (short)0, acc[mt], false, false);                     \
      if (kp < 7) {                                                                         \
        const int k2 = (2 * kp + 2) * 32;                                                   \
        b0 = *(const v16bf*)(wrow + k2 + sel * 16);                                         \
        _Pragma("unroll")                                                                   \
        for (int mt = 0; mt < 4; ++mt)                                                      \
          a0[mt] = ldA((AB) + (size_t)(mt * 16 + lh) * HID + k2 + sel * 16);                \
      }                                                                                     \
      _Pragma("unroll")                                                                     \
      for (int mt = 0; mt < 4; ++mt)                                                        \
        acc[mt] = __builtin_amdgcn_wmma_f32_16x16x32_bf16(                                  \
            false, a1[mt], false, b1, (short)0, acc[mt], false, false);                     \
    }                                                                                       \
  }

// ---------------- persistent per-layer LSTM scan ----------------
// grid = 32 WGs (blockIdx/16 = direction, blockIdx%16 = hidden-unit slice), 256 threads.
__global__ __launch_bounds__(256) void lstm_layer_kernel(
    const unsigned short* __restrict__ seq_f,   // layer input, dir 0 (bf16 [T][B][H])
    const unsigned short* __restrict__ seq_b,   // layer input, dir 1
    unsigned short* __restrict__ seq_out,       // layer-0 output base ([2][T][B][H] bf16)
    float* __restrict__ outp,                   // final output [T][B][2H] f32
    float* __restrict__ hid,                    // [4][B][H] f32
    float* __restrict__ cell,                   // [4][B][H] f32
    const unsigned short* __restrict__ wx,      // [dir*2+lay][2048][512] bf16
    const unsigned short* __restrict__ wh,
    const float* __restrict__ bias,             // [dir*2+lay][2048] f32
    unsigned short* __restrict__ hbuf,          // [2 dir][2 slab][64][512] bf16
    int* __restrict__ bar,                      // this layer's 4 ints {cnt,flg}x2dir
    int layer, int is_last) {
  extern __shared__ char smem[];
  const int tid  = threadIdx.x;
  const int lane = tid & 31;
  const int wv   = tid >> 5;          // 0..7 : N-tile owned by this wave
  const int lh   = lane & 15;
  const int sel  = lane >> 4;         // lane half selector
  const int dir  = blockIdx.x >> 4;
  const int hu0  = (blockIdx.x & 15) * NU;

  const unsigned short* seq_in = dir ? seq_b : seq_f;
  const int lsel = dir * NLAYER + layer;
  const unsigned short* WX = wx + (size_t)lsel * GATES4 * HID;
  const unsigned short* WH = wh + (size_t)lsel * GATES4 * HID;
  const float*          BI = bias + (size_t)lsel * GATES4;
  unsigned short* hb   = hbuf + (size_t)dir * 2 * BATCH * HID;
  unsigned short* so   = seq_out + (size_t)dir * T_SEQ * BATCH * HID;
  float* hid_d  = hid  + (size_t)lsel * BATCH * HID;
  float* cell_d = cell + (size_t)lsel * BATCH * HID;
  int* cnt = bar + dir * 2;
  int* flg = cnt + 1;

  unsigned short* lwx = (unsigned short*)(smem + OFF_WX);
  unsigned short* lwh = (unsigned short*)(smem + OFF_WH);
  float* gl = (float*)(smem + OFF_G);
  float* cl = (float*)(smem + OFF_C);
  float* bl = (float*)(smem + OFF_B);

  // ---- persistent weight slices via Tensor Data Mover ----
  // 8 TDM descriptors per WG: {Wx,Wh} x 4 gate blocks of 32 contiguous rows.
  if (wv == 0) {
    const unsigned lds_base = (unsigned)(uintptr_t)smem;   // LDS offset = flat[31:0]
    for (int g = 0; g < 4; ++g) {
      tdm_load_block(WX + ((size_t)(g * HID + hu0)) * HID,
                     lds_base + OFF_WX + (unsigned)g * 32u * HID * 2u);
      tdm_load_block(WH + ((size_t)(g * HID + hu0)) * HID,
                     lds_base + OFF_WH + (unsigned)g * 32u * HID * 2u);
    }
    __builtin_amdgcn_s_wait_tensorcnt((short)0);           // TENSORcnt -> 0 (wave 0)
  }
  if (tid < NC) bl[tid] = BI[(tid >> 5) * HID + hu0 + (tid & 31)];
  for (int idx = tid; idx < BATCH * NU; idx += 256) cl[idx] = 0.0f;
  for (int idx = tid; idx < BATCH * NU; idx += 256)    // zero h slab 0 (this WG's slice)
    hb[(size_t)(idx >> 5) * HID + hu0 + (idx & 31)] = 0;
  __threadfence();
  dir_barrier(cnt, flg, 1, tid);      // also publishes TDM-filled LDS to all 8 waves

  // ---- sequential scan ----
  for (int s = 0; s < T_SEQ; ++s) {
    const int t  = dir ? (T_SEQ - 1 - s) : s;
    const int rd = s & 1;            // h slab to read (double buffered)
    const int wr = rd ^ 1;           // h slab to write

    if (s + 1 < T_SEQ) {             // prefetch next timestep's activations
      int tn = dir ? (T_SEQ - 2 - s) : (s + 1);
      __builtin_prefetch(seq_in + (size_t)tn * BATCH * HID + (tid << 5), 0, 0);
    }

    v8f acc[4] = {};
    const unsigned short* ax = seq_in + (size_t)t * BATCH * HID;
    const unsigned short* ah = hb + (size_t)rd * BATCH * HID;
    GEMM_PASS(ax, lwx)               // g += x_t @ Wx_slice^T
    GEMM_PASS(ah, lwh)               // g += h_{t-1} @ Wh_slice^T

    // C layout: VGPR r -> row mt*16 + r (+8 for upper lanes), col = wv*16 + lh
#pragma unroll
    for (int mt = 0; mt < 4; ++mt)
#pragma unroll
      for (int r = 0; r < 8; ++r)
        gl[(mt * 16 + sel * 8 + r) * NC + wv * 16 + lh] = acc[mt][r];
    __syncthreads();

    // LSTM cell update for this WG's 32 hidden units x 64 batches
    for (int idx = tid; idx < BATCH * NU; idx += 256) {
      int bm = idx >> 5, uo = idx & 31;
      float iv = gl[bm * NC + uo]           + bl[uo];
      float fv = gl[bm * NC + 32 + uo]      + bl[32 + uo];
      float gv = gl[bm * NC + 64 + uo]      + bl[64 + uo];
      float ov = gl[bm * NC + 96 + uo]      + bl[96 + uo];
      float c2 = sigm(fv) * cl[idx] + sigm(iv) * tanhf(gv);
      float h2 = sigm(ov) * tanhf(c2);
      cl[idx] = c2;
      int hu = hu0 + uo;
      hb[(size_t)wr * BATCH * HID + (size_t)bm * HID + hu] = f2bf(h2);
      if (is_last)
        outp[((size_t)t * BATCH + bm) * (2 * HID) + (size_t)dir * HID + hu] = h2;
      else
        so[((size_t)t * BATCH + bm) * HID + hu] = f2bf(h2);
      if (s == T_SEQ - 1) {          // final carry -> hidden/cell outputs
        hid_d[(size_t)bm * HID + hu]  = h2;
        cell_d[(size_t)bm * HID + hu] = c2;
      }
    }
    __threadfence();
    dir_barrier(cnt, flg, s + 2, tid);
  }
}

// ---------------- launcher ----------------
extern "C" void kernel_launch(void* const* d_in, const int* in_sizes, int n_in,
                              void* d_out, int out_size, void* d_ws, size_t ws_size,
                              hipStream_t stream) {
  const float* x     = (const float*)d_in[0];
  const float* Wxh_f = (const float*)d_in[1];
  const float* bxh_f = (const float*)d_in[2];
  const float* Whh_f = (const float*)d_in[3];
  const float* bhh_f = (const float*)d_in[4];
  const float* Wxh_b = (const float*)d_in[5];
  const float* bxh_b = (const float*)d_in[6];
  const float* Whh_b = (const float*)d_in[7];
  const float* bhh_b = (const float*)d_in[8];

  char* ws = (char*)d_ws;
  unsigned short* wxbf = (unsigned short*)(ws + WS_WXBF);
  unsigned short* whbf = (unsigned short*)(ws + WS_WHBF);
  float*          bias = (float*)(ws + WS_BIAS);
  unsigned short* hbuf = (unsigned short*)(ws + WS_HBUF);
  int*            bar  = (int*)(ws + WS_BAR);
  unsigned short* xbf  = (unsigned short*)(ws + WS_XBF);
  unsigned short* seq0 = (unsigned short*)(ws + WS_SEQ0);

  float* outp = (float*)d_out;                               // [T][B][2H]
  float* hid  = outp + (size_t)T_SEQ * BATCH * 2 * HID;      // [4][B][H]
  float* cell = hid  + (size_t)2 * NLAYER * BATCH * HID;     // [4][B][H]

  const long NW = (long)NLAYER * GATES4 * HID;               // elems per direction's weight stack
  cvt_bf16_kernel<<<dim3(4096), dim3(256), 0, stream>>>(x, xbf, (long)T_SEQ * BATCH * HID);
  cvt_bf16_kernel<<<dim3(2048), dim3(256), 0, stream>>>(Wxh_f, wxbf,      NW);
  cvt_bf16_kernel<<<dim3(2048), dim3(256), 0, stream>>>(Wxh_b, wxbf + NW, NW);
  cvt_bf16_kernel<<<dim3(2048), dim3(256), 0, stream>>>(Whh_f, whbf,      NW);
  cvt_bf16_kernel<<<dim3(2048), dim3(256), 0, stream>>>(Whh_b, whbf + NW, NW);
  bias_bar_kernel<<<dim3(32), dim3(256), 0, stream>>>(bxh_f, bhh_f, bxh_b, bhh_b, bias, bar);

  (void)hipFuncSetAttribute((const void*)lstm_layer_kernel,
                            hipFuncAttributeMaxDynamicSharedMemorySize, SMEM_SZ);

  for (int layer = 0; layer < NLAYER; ++layer) {
    const unsigned short* in_f = (layer == 0) ? xbf : seq0;
    const unsigned short* in_b =
        (layer == 0) ? xbf : (seq0 + (size_t)T_SEQ * BATCH * HID);
    lstm_layer_kernel<<<dim3(2 * WG_PER_DIR), dim3(256), SMEM_SZ, stream>>>(
        in_f, in_b, seq0, outp, hid, cell, wxbf, whbf, bias, hbuf,
        bar + layer * 4, layer, (layer == NLAYER - 1) ? 1 : 0);
  }
}
